// DisLoss_45767171506360
// MI455X (gfx1250) — compile-verified, matrix operand order checked by reading, not run on hardware
//
#include <hip/hip_runtime.h>
#include <hip/hip_bf16.h>
#include <math.h>

#define NUM_CLASSES 8192
#define FEAT_DIM    256
#define BATCH       1024
#define INV_TEMP    10.0f      /* 1 / 0.1 */
#define NSPLIT      16         /* N chunks per M-tile (stride over upper triangle) */
#define NTILES      (NUM_CLASSES / 16)   /* 512 */
#define KSTEPS      (FEAT_DIM / 32)      /* 8  */

typedef __attribute__((ext_vector_type(16))) __bf16 v16bf;
typedef __attribute__((ext_vector_type(8)))  float  v8f;

union Frag { v16bf v; uint4 u[2]; };

// ---------------------------------------------------------------------------
// 1) Copy prototypes into a mutable f32 working buffer (never touch d_in).
// ---------------------------------------------------------------------------
__global__ void copy_proto_kernel(const float* __restrict__ src,
                                  float* __restrict__ dst, int n) {
    int i = blockIdx.x * blockDim.x + threadIdx.x;
    if (i < n) dst[i] = src[i];
}

// ---------------------------------------------------------------------------
// 2) Sequential EMA prototype update: one block, 256 threads (1 per feature).
//    Order over the 1024 samples must be preserved (repeated labels).
// ---------------------------------------------------------------------------
__global__ void ema_update_kernel(const float* __restrict__ feat,
                                  const int*   __restrict__ labels,
                                  float* __restrict__ protoW) {
    __shared__ float red[256];
    const int t = threadIdx.x;           // feature index 0..255
    for (int i = 0; i < BATCH; ++i) {
        const int c = labels[i];
        float v = 0.5f * protoW[c * FEAT_DIM + t] + 0.5f * feat[i * FEAT_DIM + t];
        red[t] = v * v;
        __syncthreads();
        for (int s = 128; s > 0; s >>= 1) {
            if (t < s) red[t] += red[t + s];
            __syncthreads();
        }
        const float norm = sqrtf(red[0]);
        __syncthreads();                 // red[] reused next iteration
        protoW[c * FEAT_DIM + t] = v / fmaxf(norm, 1e-12f);
    }
}

// ---------------------------------------------------------------------------
// 3) Split f32 prototypes into hi/lo bf16 (Dekker split, RNE rounding).
// ---------------------------------------------------------------------------
__device__ inline unsigned short f32_to_bf16_rne(float f) {
    unsigned int u = __float_as_uint(f);
    u += 0x7FFFu + ((u >> 16) & 1u);
    return (unsigned short)(u >> 16);
}

__global__ void split_kernel(const float* __restrict__ src,
                             unsigned short* __restrict__ phi,
                             unsigned short* __restrict__ plo, int n) {
    int i = blockIdx.x * blockDim.x + threadIdx.x;
    if (i >= n) return;
    float f  = src[i];
    unsigned short h = f32_to_bf16_rne(f);
    float hf = __uint_as_float(((unsigned int)h) << 16);
    phi[i] = h;
    plo[i] = f32_to_bf16_rne(f - hf);
}

__global__ void zero_kernel(float* __restrict__ p, int n) {
    int i = blockIdx.x * blockDim.x + threadIdx.x;
    if (i < n) p[i] = 0.0f;
}

// ---------------------------------------------------------------------------
// 4) WMMA exp(P*P^T/T) masked row-sum, upper-triangle only (symmetry).
//    One wave per block; blockIdx.x = M tile m, blockIdx.y = chunk c.
//    Wave walks n = m+c, m+c+NSPLIT, ... < NTILES.
//    Row-side sums accumulate in registers; column-side sums (the mirrored
//    tile's row contribution) are scattered per off-diagonal tile.
// ---------------------------------------------------------------------------
__device__ inline void load_frag(Frag& f, const unsigned short* __restrict__ base,
                                 int row, int kb, int lane) {
    // bf16 16x32 fragment layout: lane%16 = row, lane/16 selects K sub-halves.
    const unsigned short* p = base + row * FEAT_DIM + kb + ((lane >> 4) << 3);
    f.u[0] = *(const uint4*)(p);        // K = kb + koff .. +7
    f.u[1] = *(const uint4*)(p + 16);   // K = kb + 16 + koff .. +7
}

__global__ void __launch_bounds__(32, 1)
gemm_rowsum_kernel(const unsigned short* __restrict__ phi,
                   const unsigned short* __restrict__ plo,
                   float* __restrict__ rowSum) {
    const int lane  = threadIdx.x;                 // 0..31 (wave32)
    const int mTile = blockIdx.x;                  // 0..511
    const int chunk = blockIdx.y;                  // 0..NSPLIT-1

    const int arow  = mTile * 16 + (lane & 15);
    const int mBase = mTile * 16 + ((lane >> 4) << 3);   // row of acc[0] for this lane
    const int nlo   = lane & 15;

    // A fragments (16 rows x 256 K, hi+lo) resident: 128 VGPRs.
    Frag Ahi[KSTEPS], Alo[KSTEPS];
#pragma unroll
    for (int ks = 0; ks < KSTEPS; ++ks) {
        load_frag(Ahi[ks], phi, arow, ks * 32, lane);
        load_frag(Alo[ks], plo, arow, ks * 32, lane);
    }

    float sum[8];
#pragma unroll
    for (int r = 0; r < 8; ++r) sum[r] = 0.0f;

    for (int nt = mTile + chunk; nt < NTILES; nt += NSPLIT) {
        const int brow = nt * 16 + nlo;
        v8f acc = {0.f, 0.f, 0.f, 0.f, 0.f, 0.f, 0.f, 0.f};

        // Double-buffered B: issue next K-step's 4x b128 before current WMMAs.
        Frag Bhi[2], Blo[2];
        load_frag(Bhi[0], phi, brow, 0, lane);
        load_frag(Blo[0], plo, brow, 0, lane);
#pragma unroll
        for (int ks = 0; ks < KSTEPS; ++ks) {
            const int cur = ks & 1, nxt = cur ^ 1;
            if (ks + 1 < KSTEPS) {
                load_frag(Bhi[nxt], phi, brow, (ks + 1) * 32, lane);
                load_frag(Blo[nxt], plo, brow, (ks + 1) * 32, lane);
            }
            acc = __builtin_amdgcn_wmma_f32_16x16x32_bf16(
                false, Ahi[ks].v, false, Bhi[cur].v, (short)0, acc, false, false);
            acc = __builtin_amdgcn_wmma_f32_16x16x32_bf16(
                false, Ahi[ks].v, false, Blo[cur].v, (short)0, acc, false, false);
            acc = __builtin_amdgcn_wmma_f32_16x16x32_bf16(
                false, Alo[ks].v, false, Bhi[cur].v, (short)0, acc, false, false);
        }

        const int ncol = nt * 16 + nlo;
        float cs = 0.0f;                           // column sum (this lane's col)
#pragma unroll
        for (int r = 0; r < 8; ++r) {
            // D layout: VGPR r holds row mBase+r for this lane-half, col = ncol
            float e = ((mBase + r) == ncol) ? 0.0f : __expf(acc[r] * INV_TEMP);
            sum[r] += e;                           // row-side contribution
            cs     += e;                           // col-side partial (8 rows)
        }
        if (nt != mTile) {
            // Mirror tile (n,m): column sums of this tile are row sums of block n.
            cs += __shfl_xor(cs, 16);              // combine both 8-row halves
            if (lane < 16) atomicAdd(rowSum + ncol, cs);
        }
    }

    // Row-side: reduce over the 16 lanes of each half, one atomic per row.
#pragma unroll
    for (int r = 0; r < 8; ++r) {
        float s = sum[r];
        s += __shfl_xor(s, 1);
        s += __shfl_xor(s, 2);
        s += __shfl_xor(s, 4);
        s += __shfl_xor(s, 8);
        if ((lane & 15) == 0) atomicAdd(rowSum + mBase + r, s);
    }
}

// ---------------------------------------------------------------------------
// 5) loss = nanmean(log(rowSum / (C-1)))  (temperature/base_temperature = 1)
// ---------------------------------------------------------------------------
__global__ void finalize_kernel(const float* __restrict__ rowSum,
                                float* __restrict__ out) {
    __shared__ float ssum[256];
    __shared__ float scnt[256];
    const int t = threadIdx.x;
    float s = 0.0f, c = 0.0f;
    for (int i = t; i < NUM_CLASSES; i += 256) {
        float v = logf(rowSum[i] * (1.0f / (float)(NUM_CLASSES - 1)));
        if (!isnan(v)) { s += v; c += 1.0f; }
    }
    ssum[t] = s; scnt[t] = c;
    __syncthreads();
    for (int st = 128; st > 0; st >>= 1) {
        if (t < st) { ssum[t] += ssum[t + st]; scnt[t] += scnt[t + st]; }
        __syncthreads();
    }
    if (t == 0) out[0] = ssum[0] / scnt[0];
}

// ---------------------------------------------------------------------------
extern "C" void kernel_launch(void* const* d_in, const int* in_sizes, int n_in,
                              void* d_out, int out_size, void* d_ws, size_t ws_size,
                              hipStream_t stream) {
    const float* features   = (const float*)d_in[0];   // [1024, 256]
    const float* prototypes = (const float*)d_in[1];   // [8192, 256]
    const int*   labels     = (const int*)d_in[2];     // [1024]
    float*       out        = (float*)d_out;

    const int NP = NUM_CLASSES * FEAT_DIM;             // 2M elements

    // workspace layout
    float*          protoW = (float*)d_ws;                         // 8 MB
    unsigned short* phi    = (unsigned short*)(protoW + NP);       // 4 MB
    unsigned short* plo    = phi + NP;                             // 4 MB
    float*          rowSum = (float*)(plo + NP);                   // 32 KB

    (void)in_sizes; (void)n_in; (void)out_size; (void)ws_size;

    copy_proto_kernel<<<NP / 256, 256, 0, stream>>>(prototypes, protoW, NP);
    ema_update_kernel<<<1, 256, 0, stream>>>(features, labels, protoW);
    split_kernel<<<NP / 256, 256, 0, stream>>>(protoW, phi, plo, NP);
    zero_kernel<<<NUM_CLASSES / 256, 256, 0, stream>>>(rowSum, NUM_CLASSES);

    dim3 grid(NTILES, NSPLIT);     // upper-triangle walk, 1 wave per block
    gemm_rowsum_kernel<<<grid, 32, 0, stream>>>(phi, plo, rowSum);

    finalize_kernel<<<1, 256, 0, stream>>>(rowSum, out);
}
